// RelativeSelfAttention_81724637708737
// MI455X (gfx1250) — matrix operand order repaired and checked by reading, not compiled, not run on hardware
//
#include <hip/hip_runtime.h>

// ---------------------------------------------------------------------------
// Fused relative self-attention for gfx1250 (wave32, WMMA f32_16x16x32_f16,
// Tensor Data Mover staging for K tiles, global_prefetch for streaming tiles)
// ---------------------------------------------------------------------------

typedef __attribute__((ext_vector_type(16))) _Float16 v16h;
typedef __attribute__((ext_vector_type(8)))  _Float16 v8h;
typedef __attribute__((ext_vector_type(4)))  _Float16 v4h;
typedef __attribute__((ext_vector_type(8)))  float    v8f;
typedef __attribute__((ext_vector_type(4)))  float    v4f;
typedef __attribute__((ext_vector_type(4)))  unsigned u32x4;
typedef __attribute__((ext_vector_type(8)))  int      i32x8;
typedef __attribute__((ext_vector_type(4)))  int      i32x4;

#define D_MODEL 768
#define NHEAD   12
#define HD      64
#define T_SEQ   1024
#define BATCH   4
#define BH_TOT  (BATCH * NHEAD)

#ifndef __has_builtin
#define __has_builtin(x) 0
#endif
#if __has_builtin(__builtin_amdgcn_tensor_load_to_lds) && \
    __has_builtin(__builtin_amdgcn_s_wait_tensorcnt)
#define HAVE_TDM 1
#else
#define HAVE_TDM 0
#endif

static __device__ __forceinline__ v16h cat8(v8h lo, v8h hi) {
  return __builtin_shufflevector(lo, hi, 0,1,2,3,4,5,6,7,8,9,10,11,12,13,14,15);
}

static __device__ __forceinline__ v8f wmma_f16(v16h a, v16h b, v8f c) {
  // D = A(16x32) * B(32x16) + C, fp32 accumulate
  return __builtin_amdgcn_wmma_f32_16x16x32_f16(false, a, false, b, (short)0, c,
                                                false, false);
}

// Flat LDS pointers carry the LDS byte offset in their low 32 bits
static __device__ __forceinline__ unsigned lds_off(const void* p) {
  return (unsigned)(unsigned long long)p;
}

#if HAVE_TDM
// Issue a TDM load of a 2D f16 tile [rows x 64] (memory row stride 64 elems)
// into LDS with row padding 64 -> 72 halfs (pad 4 DWORDs every 32 DWORDs).
static __device__ __forceinline__ void tdm_load_tile64(const _Float16* gsrc,
                                                       unsigned lds_addr,
                                                       unsigned rows,
                                                       unsigned tensor_rows) {
  unsigned long long ga = (unsigned long long)gsrc;
  u32x4 g0;
  g0[0] = 1u;                                   // count=1, user mode
  g0[1] = lds_addr;                             // lds_addr [63:32]
  g0[2] = (unsigned)ga;                         // global_addr lo
  g0[3] = (unsigned)((ga >> 32) & 0x1FFFFFFu) | (2u << 30);  // addr hi | type=2

  const unsigned dim0 = HD;        // 64 elements per row
  const unsigned dim1 = tensor_rows;
  i32x8 g1;
  g1[0] = (int)((1u << 16)         // data_size = 2 bytes
                | (1u << 20)       // pad_enable
                | (4u << 22)       // pad_interval: 32 DWORDs (one 128B row)
                | (3u << 25));     // pad_amount: 4 DWORDs (16B -> stride 72h)
  g1[1] = (int)((dim0 & 0xFFFFu) << 16);                    // tensor_dim0 lo16
  g1[2] = (int)(((dim0 >> 16) & 0xFFFFu) | ((dim1 & 0xFFFFu) << 16));
  g1[3] = (int)(((dim1 >> 16) & 0xFFFFu) | (HD << 16));     // tile_dim0 = 64
  g1[4] = (int)(rows & 0xFFFFu);                            // tile_dim1, tile_dim2=0
  g1[5] = (int)HD;                                          // tensor_dim0_stride
  g1[6] = 0;
  g1[7] = 0;

  i32x4 z4 = {0, 0, 0, 0};
#if defined(__clang_major__) && (__clang_major__ >= 23)
  i32x8 z8 = {0, 0, 0, 0, 0, 0, 0, 0};
  __builtin_amdgcn_tensor_load_to_lds(g0, g1, z4, z4, z8, 0);
#else
  __builtin_amdgcn_tensor_load_to_lds(g0, g1, z4, z4, 0);
#endif
}
#endif

// ---------------------------------------------------------------------------
// Generic GEMM:  C[M,N] = A[M,K=768] * W[N,K=768]^T + bias
//   AHALF: A is f16 in workspace, else f32 (converted while staging to LDS)
//   EPI=0: split-scatter into head-separated f16 q/k/v [BH,T,64] (+bias)
//   EPI=1: f32 store to out[M,N] (+bias)
// Block tile 128x64, BK=32, 8 waves in 4x2 grid, 32x32 per wave.
// ---------------------------------------------------------------------------
template <bool AHALF, int EPI>
__global__ __launch_bounds__(256)
void gemm_wmma_kernel(const float* __restrict__ A32,
                      const _Float16* __restrict__ A16,
                      const float* __restrict__ W,
                      const float* __restrict__ bias, int N,
                      _Float16* __restrict__ qh, _Float16* __restrict__ kvk,
                      _Float16* __restrict__ kvv, float* __restrict__ outf) {
  constexpr int BM = 128, BN = 64, BK = 32, LDA = BK + 8;  // halfs, 16B-friendly
  __shared__ __attribute__((aligned(16))) _Float16 As[BM * LDA];
  __shared__ __attribute__((aligned(16))) _Float16 Bs[BN * LDA];

  const int tid  = threadIdx.x;
  const int wave = tid >> 5, lane = tid & 31;
  const int wm = wave >> 1, wn = wave & 1;  // 4x2 wave grid
  const int rowBase = blockIdx.y * BM;
  const int colBase = blockIdx.x * BN;
  const int ni = lane & 15, kh2 = lane >> 4;

  v8f acc[2][2] = {};

  for (int k0 = 0; k0 < D_MODEL; k0 += BK) {
    // prefetch next K-step tiles into cache while this step computes
    if (k0 + BK < D_MODEL) {
      int e = tid * 4;
      int r = e >> 5, c = e & 31;
      if (AHALF)
        __builtin_prefetch(A16 + (size_t)(rowBase + r) * D_MODEL + k0 + BK + c,
                           0, 1);
      else
        __builtin_prefetch(A32 + (size_t)(rowBase + r) * D_MODEL + k0 + BK + c,
                           0, 1);
      int r2 = (e >> 5) & 63;
      __builtin_prefetch(W + (size_t)(colBase + r2) * D_MODEL + k0 + BK + c, 0,
                         1);
    }
    __syncthreads();
    if (AHALF) {
#pragma unroll
      for (int it = 0; it < 2; ++it) {  // 4096 halfs, 8 per thread per it
        int e = (tid + it * 256) * 8;
        int r = e >> 5, c = e & 31;
        *(v8h*)&As[r * LDA + c] =
            *(const v8h*)(A16 + (size_t)(rowBase + r) * D_MODEL + k0 + c);
      }
    } else {
#pragma unroll
      for (int it = 0; it < 4; ++it) {  // 4096 floats, convert to f16
        int e = (tid + it * 256) * 4;
        int r = e >> 5, c = e & 31;
        v4f f = *(const v4f*)(A32 + (size_t)(rowBase + r) * D_MODEL + k0 + c);
        v4h hv;
        hv[0] = (_Float16)f[0]; hv[1] = (_Float16)f[1];
        hv[2] = (_Float16)f[2]; hv[3] = (_Float16)f[3];
        *(v4h*)&As[r * LDA + c] = hv;
      }
    }
#pragma unroll
    for (int it = 0; it < 2; ++it) {  // 2048 weight floats -> f16
      int e = (tid + it * 256) * 4;
      int r = e >> 5, c = e & 31;
      v4f f = *(const v4f*)(W + (size_t)(colBase + r) * D_MODEL + k0 + c);
      v4h hv;
      hv[0] = (_Float16)f[0]; hv[1] = (_Float16)f[1];
      hv[2] = (_Float16)f[2]; hv[3] = (_Float16)f[3];
      *(v4h*)&Bs[r * LDA + c] = hv;
    }
    __syncthreads();

    // A fragments: lane row m = ni; K = (i/8)*16 + kh2*8 + i%8
    v16h af[2], bf[2];
#pragma unroll
    for (int i = 0; i < 2; ++i) {
      int m = wm * 32 + i * 16 + ni;
      v8h lo = *(v8h*)&As[m * LDA + kh2 * 8];
      v8h hi = *(v8h*)&As[m * LDA + 16 + kh2 * 8];
      af[i] = cat8(lo, hi);
    }
    // B fragments: lane col n = ni; K = (i/8)*16 + kh2*8 + i%8
#pragma unroll
    for (int j = 0; j < 2; ++j) {
      int n = wn * 32 + j * 16 + ni;
      v8h lo = *(v8h*)&Bs[n * LDA + kh2 * 8];
      v8h hi = *(v8h*)&Bs[n * LDA + 16 + kh2 * 8];
      bf[j] = cat8(lo, hi);
    }
#pragma unroll
    for (int i = 0; i < 2; ++i)
#pragma unroll
      for (int j = 0; j < 2; ++j) acc[i][j] = wmma_f16(af[i], bf[j], acc[i][j]);
  }

  // C layout: lane col n = ni, row m = kh2*8 + e
#pragma unroll
  for (int j = 0; j < 2; ++j) {
    int col = colBase + wn * 32 + j * 16 + ni;
    float bv = bias[col];
    if (EPI == 0) {
      int part = col / D_MODEL;
      int rem  = col % D_MODEL;
      int head = rem / HD, d = rem % HD;
      _Float16* dst = (part == 0) ? qh : (part == 1) ? kvk : kvv;
#pragma unroll
      for (int i = 0; i < 2; ++i)
#pragma unroll
        for (int e = 0; e < 8; ++e) {
          int m = rowBase + wm * 32 + i * 16 + kh2 * 8 + e;
          int b = m >> 10, t = m & (T_SEQ - 1);
          dst[(((size_t)b * NHEAD + head) * T_SEQ + t) * HD + d] =
              (_Float16)(acc[i][j][e] + bv);
        }
    } else {
#pragma unroll
      for (int i = 0; i < 2; ++i)
#pragma unroll
        for (int e = 0; e < 8; ++e) {
          int m = rowBase + wm * 32 + i * 16 + kh2 * 8 + e;
          outf[(size_t)m * N + col] = acc[i][j][e] + bv;
        }
    }
  }
}

// ---------------------------------------------------------------------------
// Flash attention with relative-position bias.
// Grid: (BH=48, T/128). 8 waves, each owns 16 t-rows; S streamed in 64-chunks.
// K tiles staged by the Tensor Data Mover (TDM) with LDS row padding; V tiles
// staged cooperatively and transposed for B-fragments.
// rel bias: P[m][j] = q[t0+m] . rel_emb[j]  (33 distinct columns), gathered
// per score element with j = clip(s-t,-16,16)+16.
// ---------------------------------------------------------------------------
__global__ __launch_bounds__(256)
void attn_wmma_kernel(const _Float16* __restrict__ qh,
                      const _Float16* __restrict__ kh,
                      const _Float16* __restrict__ vh,
                      const float* __restrict__ rel,
                      _Float16* __restrict__ ao) {
  constexpr int SC = 64, LDK = HD + 8;  // 72 halfs, keeps 16B alignment
  __shared__ __attribute__((aligned(16))) _Float16 Ks[SC * LDK];  // [s][d]
  __shared__ __attribute__((aligned(16))) _Float16 Vt[HD * LDK];  // [d][s]
  __shared__ float Ps[8][16][36];                                  // rel table
  __shared__ __attribute__((aligned(16))) _Float16 Pr[8][16 * LDK];  // probs

  const int tid = threadIdx.x, wave = tid >> 5, lane = tid & 31;
  const int ni = lane & 15, kh2 = lane >> 4;
  const int bh = blockIdx.x, b = bh / NHEAD, h = bh % NHEAD;
  const int t0 = blockIdx.y * 128 + wave * 16;

  const _Float16* qg = qh + ((size_t)bh * T_SEQ + t0) * HD;
  const _Float16* kb = kh + (size_t)bh * T_SEQ * HD;
  const _Float16* vb = vh + (size_t)bh * T_SEQ * HD;

  // q A-fragments live in registers for the whole kernel (K = d = 64 -> 2)
  v16h qa[2];
#pragma unroll
  for (int g = 0; g < 2; ++g) {
    v8h lo = *(const v8h*)(qg + ni * HD + g * 32 + kh2 * 8);
    v8h hi = *(const v8h*)(qg + ni * HD + g * 32 + 16 + kh2 * 8);
    qa[g] = cat8(lo, hi);
  }

  // per-wave rel-pos dot table
  for (int idx = lane; idx < 16 * 33; idx += 32) {
    int m = idx / 33, j = idx % 33;
    const _Float16* qrow = qg + m * HD;
    const float* rrow = rel + j * HD;
    float s = 0.f;
    for (int d = 0; d < HD; ++d) s += (float)qrow[d] * rrow[d];
    Ps[wave][m][j] = s;
  }

  v8f o[4] = {};
  float mrun[8], lrun[8];
#pragma unroll
  for (int e = 0; e < 8; ++e) { mrun[e] = -3.0e38f; lrun[e] = 0.f; }

  const unsigned ks_lds = lds_off(&Ks[0]);

  for (int sc = 0; sc < T_SEQ; sc += SC) {
    // prefetch next V chunk while this one computes
    if (sc + SC < T_SEQ)
      __builtin_prefetch(vb + (size_t)(sc + SC + (tid >> 2)) * HD, 0, 1);
    __syncthreads();
#if HAVE_TDM
    // K chunk via Tensor Data Mover (row pad 64 -> 72 halfs done by TDM)
    if (wave == 0) tdm_load_tile64(kb + (size_t)sc * HD, ks_lds, SC, T_SEQ);
#else
#pragma unroll
    for (int it = 0; it < 2; ++it) {
      int e = (tid + it * 256) * 8;
      int r = e >> 6, c = e & 63;
      *(v8h*)&Ks[r * LDK + c] = *(const v8h*)(kb + (size_t)(sc + r) * HD + c);
    }
#endif
    // V chunk cooperative load + transpose into Vt[d][s]
#pragma unroll
    for (int it = 0; it < 2; ++it) {  // 4096 halfs
      int e = (tid + it * 256) * 8;
      int r = e >> 6, c = e & 63;
      v8h vv = *(const v8h*)(vb + (size_t)(sc + r) * HD + c);
#pragma unroll
      for (int u = 0; u < 8; ++u) Vt[(c + u) * LDK + r] = vv[u];
    }
#if HAVE_TDM
    if (wave == 0) __builtin_amdgcn_s_wait_tensorcnt((short)0);
#endif
    __syncthreads();

    // scores: q(16x64) . K^T -> 4 tiles of 16x16
    float sv[4][8];
#pragma unroll
    for (int jt = 0; jt < 4; ++jt) {
      v8f s8 = {};
#pragma unroll
      for (int g = 0; g < 2; ++g) {
        v8h lo = *(v8h*)&Ks[(jt * 16 + ni) * LDK + g * 32 + kh2 * 8];
        v8h hi = *(v8h*)&Ks[(jt * 16 + ni) * LDK + g * 32 + 16 + kh2 * 8];
        s8 = wmma_f16(qa[g], cat8(lo, hi), s8);
      }
#pragma unroll
      for (int e = 0; e < 8; ++e) sv[jt][e] = s8[e];
    }

    // rel bias + scale, then row max (rows live across 16 lanes of a half)
    float rmax[8];
#pragma unroll
    for (int e = 0; e < 8; ++e) rmax[e] = -3.0e38f;
#pragma unroll
    for (int jt = 0; jt < 4; ++jt)
#pragma unroll
      for (int e = 0; e < 8; ++e) {
        int s = sc + jt * 16 + ni;
        int m = kh2 * 8 + e;
        int dj = s - (t0 + m);
        dj = (dj < -16) ? -16 : (dj > 16) ? 16 : dj;
        float v = (sv[jt][e] + Ps[wave][m][dj + 16]) * 0.125f;
        sv[jt][e] = v;
        rmax[e] = fmaxf(rmax[e], v);
      }
#pragma unroll
    for (int off = 1; off < 16; off <<= 1)
#pragma unroll
      for (int e = 0; e < 8; ++e)
        rmax[e] = fmaxf(rmax[e], __shfl_xor(rmax[e], off, 32));

    float alpha[8], rsum[8];
#pragma unroll
    for (int e = 0; e < 8; ++e) {
      float mn = fmaxf(mrun[e], rmax[e]);
      alpha[e] = __expf(mrun[e] - mn);
      mrun[e]  = mn;
      rsum[e]  = 0.f;
    }
#pragma unroll
    for (int jt = 0; jt < 4; ++jt)
#pragma unroll
      for (int e = 0; e < 8; ++e) {
        float p = __expf(sv[jt][e] - mrun[e]);
        sv[jt][e] = p;
        rsum[e] += p;
      }
#pragma unroll
    for (int off = 1; off < 16; off <<= 1)
#pragma unroll
      for (int e = 0; e < 8; ++e) rsum[e] += __shfl_xor(rsum[e], off, 32);
#pragma unroll
    for (int e = 0; e < 8; ++e) lrun[e] = lrun[e] * alpha[e] + rsum[e];
#pragma unroll
    for (int dt = 0; dt < 4; ++dt)
#pragma unroll
      for (int e = 0; e < 8; ++e) o[dt][e] *= alpha[e];

    // re-layout probs C-fragment -> A-fragment via per-wave LDS scratch
    _Float16* pw = &Pr[wave][0];
#pragma unroll
    for (int jt = 0; jt < 4; ++jt)
#pragma unroll
      for (int e = 0; e < 8; ++e)
        pw[(kh2 * 8 + e) * LDK + jt * 16 + ni] = (_Float16)sv[jt][e];

    v16h pa[2];
#pragma unroll
    for (int g = 0; g < 2; ++g) {
      v8h lo = *(v8h*)&pw[ni * LDK + g * 32 + kh2 * 8];
      v8h hi = *(v8h*)&pw[ni * LDK + g * 32 + 16 + kh2 * 8];
      pa[g] = cat8(lo, hi);
    }
    // O += probs(16x64) . V(64x64), B-fragments from transposed Vt
#pragma unroll
    for (int dt = 0; dt < 4; ++dt)
#pragma unroll
      for (int g = 0; g < 2; ++g) {
        v8h lo = *(v8h*)&Vt[(dt * 16 + ni) * LDK + g * 32 + kh2 * 8];
        v8h hi = *(v8h*)&Vt[(dt * 16 + ni) * LDK + g * 32 + 16 + kh2 * 8];
        o[dt] = wmma_f16(pa[g], cat8(lo, hi), o[dt]);
      }
  }

  // normalize and store [B,T,D] f16 for the output projection
#pragma unroll
  for (int dt = 0; dt < 4; ++dt)
#pragma unroll
    for (int e = 0; e < 8; ++e) {
      int t = t0 + kh2 * 8 + e;
      int col = h * HD + dt * 16 + ni;
      ao[((size_t)b * T_SEQ + t) * D_MODEL + col] =
          (_Float16)(o[dt][e] / lrun[e]);
    }
}

// ---------------------------------------------------------------------------
extern "C" void kernel_launch(void* const* d_in, const int* in_sizes, int n_in,
                              void* d_out, int out_size, void* d_ws,
                              size_t ws_size, hipStream_t stream) {
  (void)in_sizes; (void)n_in; (void)out_size; (void)ws_size;
  const float* x     = (const float*)d_in[0];  // [4,1024,768]
  const float* qkv_w = (const float*)d_in[1];  // [2304,768]
  const float* qkv_b = (const float*)d_in[2];  // [2304]
  const float* out_w = (const float*)d_in[3];  // [768,768]
  const float* out_b = (const float*)d_in[4];  // [768]
  const float* rel   = (const float*)d_in[5];  // [33,64]
  float* out = (float*)d_out;                  // [4,1024,768]

  const size_t qkv_elems = (size_t)BH_TOT * T_SEQ * HD;  // 3,145,728
  _Float16* qh = (_Float16*)d_ws;
  _Float16* kh = qh + qkv_elems;
  _Float16* vh = kh + qkv_elems;
  _Float16* ao = vh + qkv_elems;  // [4096,768] f16

  const int M = BATCH * T_SEQ;  // 4096

  // 1) QKV projection, scatter into head-separated f16 q/k/v
  gemm_wmma_kernel<false, 0><<<dim3((3 * D_MODEL) / 64, M / 128), 256, 0,
                               stream>>>(x, nullptr, qkv_w, qkv_b, 3 * D_MODEL,
                                         qh, kh, vh, nullptr);

  // 2) fused relative-position flash attention (TDM-staged K tiles)
  attn_wmma_kernel<<<dim3(BH_TOT, T_SEQ / 128), 256, 0, stream>>>(qh, kh, vh,
                                                                  rel, ao);

  // 3) output projection -> fp32 d_out
  gemm_wmma_kernel<true, 1><<<dim3(D_MODEL / 64, M / 128), 256, 0, stream>>>(
      nullptr, ao, out_w, out_b, D_MODEL, nullptr, nullptr, nullptr, out);
}
